// VulDetectionModel_8735963480511
// MI455X (gfx1250) — compile-verified, matrix operand order checked by reading, not run on hardware
//
#include <hip/hip_runtime.h>
#include <hip/hip_bf16.h>
#include <math.h>

// ---------------------------------------------------------------------------
// Model dims
#define HD 256
#define NHEADS 4
#define HC 64
#define GSTEPS 6
#define TOPKN 12
#define BB 8
#define KK 64
#define MM 32
#define LL 512
#define LNL 16
#define NAST 50000
#define NCFG 20000
#define NPDG 20000
#define E_TOK 150000
#define E_STM 100000
#define E_BLK 80000
#define E_CFG 60000
#define E_PDG 80000

// activation codes
#define ACT_NONE 0
#define ACT_RELU 1
#define ACT_GELU 2
#define ACT_ELU  3

typedef __bf16 bf16_t;
typedef __attribute__((ext_vector_type(16))) __bf16 v16bf;
typedef __attribute__((ext_vector_type(4)))  __bf16 v4bf;
typedef __attribute__((ext_vector_type(8)))  float  v8f;
typedef __attribute__((ext_vector_type(4)))  float  v4f;

__device__ __forceinline__ float apply_act(int act, float v) {
    switch (act) {
        case ACT_RELU: return v > 0.f ? v : 0.f;
        case ACT_GELU: return 0.5f * v * (1.f + erff(v * 0.70710678118654752f));
        case ACT_ELU:  return v > 0.f ? v : (__expf(v) - 1.f);
        default:       return v;
    }
}

// ---------------------------------------------------------------------------
// WMMA bf16 GEMM:  C[M,N] = act(A[M,K] @ W[K,N] + bias)
// A row-major stride lda, W row-major stride N, C row-major stride ldc.
// Requirements (all call sites satisfy): Kd % 64 == 0, lda % 4 == 0, N % 4 == 0.
// Block tile 128x128, K staged 64 at a time in LDS as bf16 (2 WMMA k-steps).
// 256 threads = 8 waves; wave (wm,wn) computes a 32x64 sub-tile (2x4 WMMA tiles).
#define BM 128
#define BN 128
#define BKK 64

__global__ __launch_bounds__(256) void gemm_bf16_wmma(
    const float* __restrict__ A, const float* __restrict__ W,
    const float* __restrict__ bias, float* __restrict__ C,
    int M, int N, int Kd, int lda, int ldc, int act)
{
    __shared__ bf16_t sA[BM][BKK];
    __shared__ bf16_t sB[BN][BKK];   // sB[n][k]

    const int tid  = threadIdx.x;
    const int wave = tid >> 5;
    const int lane = tid & 31;
    const int wm = wave & 3;   // 4 wave-rows of 32
    const int wn = wave >> 2;  // 2 wave-cols of 64
    const int block_m = blockIdx.y * BM;
    const int block_n = blockIdx.x * BN;

    const int row16 = lane & 15;
    const int kg    = lane >> 4;   // 0/1
    const int kb0   = kg * 8;      // elems 0..7 read k = kb0 + e
    const int kb1   = 16 + kg * 8; // elems 8..15 read k = kb1 + (e-8)

    v8f acc[2][4];
#pragma unroll
    for (int im = 0; im < 2; ++im)
#pragma unroll
        for (int in = 0; in < 4; ++in)
#pragma unroll
            for (int e = 0; e < 8; ++e) acc[im][in][e] = 0.f;

    for (int k0 = 0; k0 < Kd; k0 += BKK) {
        // ---- stage A tile: 128x64 floats -> bf16, branch-free (row clamp) --
        // 8192 elements / 256 threads = 8 x float4 per thread.
#pragma unroll
        for (int i = 0; i < 8; ++i) {
            int v = tid + i * 256;       // vec4 index, 0..2047
            int r = v >> 4;              // 0..127
            int c = (v & 15) * 4;        // 0..60, multiple of 4
            int gr = block_m + r;
            int grc = gr < M ? gr : (M - 1);
            const float* ap = A + (long)grc * lda + (k0 + c);
            if (i == 0 && k0 + BKK < Kd) __builtin_prefetch(ap + BKK, 0, 1);
            v4f f = *(const v4f*)ap;
            if (gr >= M) { f[0] = 0.f; f[1] = 0.f; f[2] = 0.f; f[3] = 0.f; }
            v4bf h;
            h[0] = (bf16_t)f[0]; h[1] = (bf16_t)f[1];
            h[2] = (bf16_t)f[2]; h[3] = (bf16_t)f[3];
            *(v4bf*)&sA[r][c] = h;
        }
        // ---- stage W tile: 64x128 floats -> bf16 transposed, branch-free ---
#pragma unroll
        for (int i = 0; i < 8; ++i) {
            int v = tid + i * 256;       // vec4 index, 0..2047
            int kr = v >> 5;             // 0..63
            int n4 = (v & 31) * 4;       // 0..124, multiple of 4
            int gn = block_n + n4;
            int gnc = gn < N ? gn : 0;
            const float* wp = W + (long)(k0 + kr) * N + gnc;
            v4f f = *(const v4f*)wp;
            if (gn >= N) { f[0] = 0.f; f[1] = 0.f; f[2] = 0.f; f[3] = 0.f; }
            sB[n4 + 0][kr] = (bf16_t)f[0];
            sB[n4 + 1][kr] = (bf16_t)f[1];
            sB[n4 + 2][kr] = (bf16_t)f[2];
            sB[n4 + 3][kr] = (bf16_t)f[3];
        }
        __syncthreads();

#pragma unroll
        for (int ks = 0; ks < 2; ++ks) {
            const int kb = ks * 32;
            v16bf af[2], bfr[4];
#pragma unroll
            for (int im = 0; im < 2; ++im) {
                int r = wm * 32 + im * 16 + row16;
#pragma unroll
                for (int e = 0; e < 8; ++e) {
                    af[im][e]     = sA[r][kb + kb0 + e];
                    af[im][8 + e] = sA[r][kb + kb1 + e];
                }
            }
#pragma unroll
            for (int in = 0; in < 4; ++in) {
                int n = wn * 64 + in * 16 + row16;
#pragma unroll
                for (int e = 0; e < 8; ++e) {
                    bfr[in][e]     = sB[n][kb + kb0 + e];
                    bfr[in][8 + e] = sB[n][kb + kb1 + e];
                }
            }
#pragma unroll
            for (int im = 0; im < 2; ++im)
#pragma unroll
                for (int in = 0; in < 4; ++in)
                    acc[im][in] = __builtin_amdgcn_wmma_f32_16x16x32_bf16(
                        false, af[im], false, bfr[in], (short)0, acc[im][in], false, false);
        }
        __syncthreads();
    }

    const int lrow = lane >> 4;
    const int lcol = lane & 15;
#pragma unroll
    for (int im = 0; im < 2; ++im)
#pragma unroll
        for (int in = 0; in < 4; ++in) {
            int col = block_n + wn * 64 + in * 16 + lcol;
            if (col >= N) continue;
            float bv = bias ? bias[col] : 0.f;
#pragma unroll
            for (int r = 0; r < 8; ++r) {
                int row = block_m + wm * 32 + im * 16 + lrow * 8 + r;
                if (row < M) {
                    float v = acc[im][in][r] + bv;
                    C[(long)row * ldc + col] = apply_act(act, v);
                }
            }
        }
}

// ---------------------------------------------------------------------------
// Utility kernels
__global__ void fill_f(float* p, float v, long n) {
    long i = (long)blockIdx.x * blockDim.x + threadIdx.x;
    if (i < n) p[i] = v;
}
__global__ void fill_u(unsigned* p, unsigned v, long n) {
    long i = (long)blockIdx.x * blockDim.x + threadIdx.x;
    if (i < n) p[i] = v;
}
__global__ void copy_f(float* dst, const float* src, long n) {
    long i = (long)blockIdx.x * blockDim.x + threadIdx.x;
    if (i < n) dst[i] = src[i];
}
__global__ void copy_block(float* dst, int dld, int doff, const float* src, int D, long total) {
    long i = (long)blockIdx.x * blockDim.x + threadIdx.x;
    if (i >= total) return;
    long r = i / D; int d = (int)(i - r * D);
    dst[r * dld + doff + d] = src[i];
}
__global__ void mask_to_float(const unsigned char* m, float* o, int n) {
    int i = blockIdx.x * blockDim.x + threadIdx.x;
    if (i < n) o[i] = m[i] ? 1.f : 0.f;
}
__global__ void init_rows_bias(float* out, const float* bias, long total, int D) {
    long i = (long)blockIdx.x * blockDim.x + threadIdx.x;
    if (i >= total) return;
    out[i] = bias[i % D];
}

// ---------------------------------------------------------------------------
// GCN
__global__ void deg_accum(const int* dst, int E, float* deg) {
    int e = blockIdx.x * blockDim.x + threadIdx.x;
    if (e < E) atomicAdd(&deg[dst[e]], 1.f);
}
__global__ void dinv_finish(float* deg, int N) {
    int i = blockIdx.x * blockDim.x + threadIdx.x;
    if (i < N) deg[i] = rsqrtf(fmaxf(deg[i], 1.f));
}
__global__ void gcn_self(const float* __restrict__ xw, const float* __restrict__ dinv,
                         const float* __restrict__ bias, float* __restrict__ out, long total) {
    long i = (long)blockIdx.x * blockDim.x + threadIdx.x;
    if (i >= total) return;
    long n = i >> 8; int d = (int)(i & 255);
    float di = dinv[n];
    out[i] = bias[d] + xw[i] * di * di;
}
__global__ void gcn_scatter(const float* __restrict__ xw, const int* __restrict__ src,
                            const int* __restrict__ dst, const float* __restrict__ dinv,
                            float* __restrict__ out) {
    int e = blockIdx.x, d = threadIdx.x;
    int s = src[e], t = dst[e];
    float w = dinv[s] * dinv[t];
    atomicAdd(&out[(long)t * HD + d], xw[(long)s * HD + d] * w);
}
// plain segment-sum scatter (GGNN messages)
__global__ void seg_sum_scatter(const float* __restrict__ val, const int* __restrict__ src,
                                const int* __restrict__ dst, float* __restrict__ out) {
    int e = blockIdx.x, d = threadIdx.x;
    atomicAdd(&out[(long)dst[e] * HD + d], val[(long)src[e] * HD + d]);
}

// ---------------------------------------------------------------------------
// LayerNorm over rows with optional residual and pre/post activation
__global__ void row_ln(const float* __restrict__ x, const float* __restrict__ res,
                       float* __restrict__ out, const float* __restrict__ g,
                       const float* __restrict__ b, int D, int pre_act, int post_act) {
    int r = blockIdx.x, t = threadIdx.x;
    long o = (long)r * D + t;
    float v = x[o] + (res ? res[o] : 0.f);
    v = apply_act(pre_act, v);
    __shared__ float sred[256];
    __shared__ float smean, svar;
    sred[t] = v; __syncthreads();
    for (int s = blockDim.x >> 1; s > 0; s >>= 1) {
        if (t < s) sred[t] += sred[t + s];
        __syncthreads();
    }
    if (t == 0) smean = sred[0] / (float)D;
    __syncthreads();
    float dv = v - smean;
    sred[t] = dv * dv; __syncthreads();
    for (int s = blockDim.x >> 1; s > 0; s >>= 1) {
        if (t < s) sred[t] += sred[t + s];
        __syncthreads();
    }
    if (t == 0) svar = sred[0] / (float)D;
    __syncthreads();
    float y = dv * rsqrtf(svar + 1e-5f) * g[t] + b[t];
    out[o] = apply_act(post_act, y);
}

// ---------------------------------------------------------------------------
// GGNN GRU gate
__global__ void gru_gate(const float* __restrict__ gi, const float* __restrict__ gh,
                         const float* __restrict__ h, float* __restrict__ hn, long total) {
    long i = (long)blockIdx.x * blockDim.x + threadIdx.x;
    if (i >= total) return;
    long n = i >> 8; int d = (int)(i & 255);
    const float* gp = gi + n * 768;
    const float* hp = gh + n * 768;
    float r  = 1.f / (1.f + __expf(-(gp[d] + hp[d])));
    float z  = 1.f / (1.f + __expf(-(gp[256 + d] + hp[256 + d])));
    float ng = tanhf(gp[512 + d] + r * hp[512 + d]);
    hn[i] = (1.f - z) * ng + z * h[i];
}

// ---------------------------------------------------------------------------
// GAT
__device__ __forceinline__ unsigned fenc(float f) {
    unsigned u = __float_as_uint(f);
    return (u & 0x80000000u) ? ~u : (u | 0x80000000u);
}
__device__ __forceinline__ float fdec(unsigned u) {
    unsigned v = (u & 0x80000000u) ? (u & 0x7fffffffu) : ~u;
    return __uint_as_float(v);
}
__global__ void gat_coef(const float* __restrict__ hh, const float* __restrict__ aw_s,
                         const float* __restrict__ aw_d, float* __restrict__ a_src,
                         float* __restrict__ a_dst, int N) {
    int i = blockIdx.x * blockDim.x + threadIdx.x;
    if (i >= N * NHEADS) return;
    int n = i >> 2, h = i & 3;
    const float* p = hh + (long)n * HD + h * HC;
    float s1 = 0.f, s2 = 0.f;
#pragma unroll 8
    for (int d = 0; d < HC; ++d) { float v = p[d]; s1 += v * aw_s[h * HC + d]; s2 += v * aw_d[h * HC + d]; }
    a_src[i] = s1; a_dst[i] = s2;
}
__device__ __forceinline__ void gat_edge(int e, const int* src, const int* dst, int E,
                                         int& s, int& t) {
    if (e < E) { s = src[e]; t = dst[e]; } else { s = t = e - E; }
}
__global__ void gat_edge_max(const int* src, const int* dst, const float* a_src,
                             const float* a_dst, unsigned* mx, int E, int N) {
    int i = blockIdx.x * blockDim.x + threadIdx.x;
    if (i >= (E + N) * NHEADS) return;
    int e = i >> 2, h = i & 3, s, t;
    gat_edge(e, src, dst, E, s, t);
    float x = a_src[s * 4 + h] + a_dst[t * 4 + h];
    x = x > 0.f ? x : 0.2f * x;
    atomicMax(&mx[t * 4 + h], fenc(x));
}
__global__ void gat_edge_exp(const int* src, const int* dst, const float* a_src,
                             const float* a_dst, const unsigned* mx, float* ebuf,
                             float* denom, int E, int N) {
    int i = blockIdx.x * blockDim.x + threadIdx.x;
    if (i >= (E + N) * NHEADS) return;
    int e = i >> 2, h = i & 3, s, t;
    gat_edge(e, src, dst, E, s, t);
    float x = a_src[s * 4 + h] + a_dst[t * 4 + h];
    x = x > 0.f ? x : 0.2f * x;
    float ee = __expf(x - fdec(mx[t * 4 + h]));
    ebuf[i] = ee;
    atomicAdd(&denom[t * 4 + h], ee);
}
__global__ void gat_scatter(const float* __restrict__ hh, const int* src, const int* dst,
                            const float* __restrict__ ebuf, const float* __restrict__ denom,
                            float* __restrict__ out, int E, int N) {
    long i = (long)blockIdx.x * blockDim.x + threadIdx.x;
    if (i >= (long)(E + N) * NHEADS * HC) return;
    int d = (int)(i & 63);
    int h = (int)((i >> 6) & 3);
    int e = (int)(i >> 8);
    int s, t; gat_edge(e, src, dst, E, s, t);
    float alpha = ebuf[e * 4 + h] / (denom[t * 4 + h] + 1e-16f);
    atomicAdd(&out[(long)t * HD + h * HC + d], hh[(long)s * HD + h * HC + d] * alpha);
}

// ---------------------------------------------------------------------------
// Transformer: pad mask + attention
__global__ void pad_mask_kernel(const float* __restrict__ lines, int* pad, int total) {
    int i = blockIdx.x * blockDim.x + threadIdx.x;
    if (i >= total) return;
    const float* p = lines + (long)i * HD;
    float s = 0.f;
    for (int d = 0; d < HD; ++d) s += p[d] * p[d];
    pad[i] = (sqrtf(s) < 1e-6f) ? 1 : 0;
}
__global__ void pad_fix_kernel(int* pad, int L) {
    if (threadIdx.x != 0) return;
    int* p = pad + blockIdx.x * L;
    for (int l = 0; l < L; ++l) if (!p[l]) return;
    p[0] = 0;
}
__global__ __launch_bounds__(128) void attn_kernel(const float* __restrict__ qkv,
                                                   const int* __restrict__ pad,
                                                   float* __restrict__ o, int L) {
    int l = blockIdx.x, h = blockIdx.y, b = blockIdx.z;
    int t = threadIdx.x;
    const float* base = qkv + (long)b * L * 768;
    __shared__ float sq[HC];
    __shared__ float ss[LL];
    __shared__ float red[128];
    if (t < HC) sq[t] = base[(long)l * 768 + h * HC + t];
    __syncthreads();
    for (int m = t; m < L; m += 128) {
        const float* kp = base + (long)m * 768 + 256 + h * HC;
        float s = 0.f;
#pragma unroll 8
        for (int d = 0; d < HC; ++d) s += sq[d] * kp[d];
        s *= 0.125f;
        if (pad[b * L + m]) s = -1e9f;
        ss[m] = s;
    }
    __syncthreads();
    float mx = -3e38f;
    for (int m = t; m < L; m += 128) mx = fmaxf(mx, ss[m]);
    red[t] = mx; __syncthreads();
    for (int s = 64; s > 0; s >>= 1) { if (t < s) red[t] = fmaxf(red[t], red[t + s]); __syncthreads(); }
    mx = red[0]; __syncthreads();
    float sum = 0.f;
    for (int m = t; m < L; m += 128) { float e = __expf(ss[m] - mx); ss[m] = e; sum += e; }
    red[t] = sum; __syncthreads();
    for (int s = 64; s > 0; s >>= 1) { if (t < s) red[t] += red[t + s]; __syncthreads(); }
    sum = red[0]; __syncthreads();
    if (t < HC) {
        float acc = 0.f;
        for (int m = 0; m < L; ++m) acc += ss[m] * base[(long)m * 768 + 512 + h * HC + t];
        o[((long)(b * L + l)) * HD + h * HC + t] = acc / sum;
    }
}

// ---------------------------------------------------------------------------
// Pooling
__global__ void region_pool(const float* __restrict__ feat, const int* __restrict__ nodes,
                            const unsigned char* __restrict__ mask, float* __restrict__ out,
                            int Mm, int out_ld, int out_off) {
    int r = blockIdx.x, d = threadIdx.x;
    float s = 0.f; int cnt = 0;
    for (int j = 0; j < Mm; ++j) {
        if (mask[r * Mm + j]) { s += feat[(long)nodes[r * Mm + j] * HD + d]; ++cnt; }
    }
    out[(long)r * out_ld + out_off + d] = s / (float)(cnt > 0 ? cnt : 1);
}
__global__ void line_pool(const float* __restrict__ sem, const int* __restrict__ idx,
                          const unsigned char* __restrict__ mask, float* __restrict__ out,
                          int Kk, int LNn, int L, int out_ld, int out_off) {
    int r = blockIdx.x, d = threadIdx.x;
    int b = r / Kk;
    float s = 0.f; int cnt = 0;
    for (int j = 0; j < LNn; ++j) {
        if (mask[r * LNn + j]) {
            int li = idx[r * LNn + j];
            li = li < 0 ? 0 : (li > L - 1 ? L - 1 : li);
            s += sem[((long)b * L + li) * HD + d]; ++cnt;
        }
    }
    out[(long)r * out_ld + out_off + d] = s / (float)(cnt > 0 ? cnt : 1);
}

// ---------------------------------------------------------------------------
// Heads / top-k
__global__ void dot64(const float* __restrict__ x, const float* __restrict__ w,
                      const float* __restrict__ bsc, float* __restrict__ out, int R) {
    int r = blockIdx.x, t = threadIdx.x;
    __shared__ float red[64];
    red[t] = x[(long)r * 64 + t] * w[t];
    __syncthreads();
    for (int s = 32; s > 0; s >>= 1) { if (t < s) red[t] += red[t + s]; __syncthreads(); }
    if (t == 0) out[r] = red[0] + bsc[0];
}
__global__ __launch_bounds__(256) void topk_kernel(const float* __restrict__ feat,
                                                   const float* __restrict__ logits,
                                                   const unsigned char* __restrict__ rmask,
                                                   float* __restrict__ gfeat,
                                                   float* __restrict__ oidx,
                                                   float* __restrict__ osc) {
    int b = blockIdx.x, t = threadIdx.x;
    __shared__ float sc[KK];
    __shared__ int ch[TOPKN];
    __shared__ float wsel[TOPKN];
    __shared__ float sden;
    if (t < KK) {
        float lg = logits[b * KK + t];
        sc[t] = rmask[b * KK + t] ? 1.f / (1.f + __expf(-lg)) : -1e9f;
    }
    __syncthreads();
    if (t == 0) {
        float den = 0.f;
        for (int j = 0; j < TOPKN; ++j) {
            int bi = 0; float bv = -3e38f;
            for (int i = 0; i < KK; ++i) if (sc[i] > bv) { bv = sc[i]; bi = i; }
            sc[bi] = -3e38f;
            bool valid = bv > -1e8f;
            ch[j] = valid ? bi : -1;
            float w = valid ? fmaxf(bv, 0.f) : 0.f;
            wsel[j] = w; den += w;
            oidx[b * TOPKN + j] = valid ? (float)bi : -1.f;
            osc[b * TOPKN + j]  = valid ? bv : 0.f;
        }
        sden = fmaxf(den, 1e-6f);
    }
    __syncthreads();
    float acc = 0.f;
    for (int j = 0; j < TOPKN; ++j)
        if (ch[j] >= 0) acc += feat[((long)b * KK + ch[j]) * HD + t] * wsel[j];
    gfeat[b * HD + t] = acc / sden;
}

// ---------------------------------------------------------------------------
// Input index map (setup_inputs dict order; params pytree = sorted dict keys)
enum {
    IN_AST_X = 0, IN_CFG_X, IN_PDG_X, IN_LINES,
    IN_TOKEN_EI, IN_STMT_EI, IN_BLOCK_EI, IN_CFG_EI, IN_PDG_EI,
    IN_R_AST, IN_R_AST_M, IN_R_CFG, IN_R_CFG_M, IN_R_PDG, IN_R_PDG_M,
    IN_R_LINE, IN_R_LINE_M, IN_R_MASK,
    P_AST_BW, P_AST_BB, P_AST_BBT, P_AST_BG,
    P_AST_O1W, P_AST_O1B, P_AST_O2W, P_AST_O2B,
    P_AST_SW, P_AST_SB, P_AST_SBT, P_AST_SG,
    P_AST_TW, P_AST_TB, P_AST_TBT, P_AST_TG,
    P_CFG_WHH, P_CFG_WIH, P_CFG_B, P_CFG_BHH, P_CFG_BIH, P_CFG_G, P_CFG_W,
    P_CLS_W1, P_CLS_W2, P_CLS_W3, P_CLS_B1, P_CLS_B2, P_CLS_B3,
    P_CLS_BT1, P_CLS_BT2, P_CLS_G1, P_CLS_G2,
    P_FUSE_W, P_FUSE_B, P_FUSE_BT, P_FUSE_G,
    P_PDG_G1_W, P_PDG_G1_AD, P_PDG_G1_AS, P_PDG_G1_BIAS,
    P_PDG_G2_W, P_PDG_G2_AD, P_PDG_G2_AS, P_PDG_G2_BIAS,
    P_PDG_LN1_B, P_PDG_LN1_G, P_PDG_LN2_B, P_PDG_LN2_G,
    P_SC_W1, P_SC_W2, P_SC_W3, P_SC_B1, P_SC_B2, P_SC_B3,
    P_SC_BT1, P_SC_BT2, P_SC_G1, P_SC_G2,
    P_SEM0  // per layer (x3): W1,W2,Wo,Wqkv,b1,b2,bo,bqkv,ln1_b,ln1_g,ln2_b,ln2_g
};
enum { SW1 = 0, SW2, SWO, SWQKV, SB1, SB2, SBO, SBQKV, SLN1B, SLN1G, SLN2B, SLN2G };

extern "C" void kernel_launch(void* const* d_in, const int* in_sizes, int n_in,
                              void* d_out, int out_size, void* d_ws, size_t ws_size,
                              hipStream_t stream) {
    (void)in_sizes; (void)n_in; (void)out_size; (void)ws_size;
    auto PF = [&](int i) { return (const float*)d_in[i]; };
    auto PI = [&](int i) { return (const int*)d_in[i]; };
    auto PB = [&](int i) { return (const unsigned char*)d_in[i]; };

    float* out = (float*)d_out;
    float* o_logits = out;                       // [8,64]
    float* o_mask   = out + 512;                 // [8,64]
    float* o_feat   = out + 1024;                // [8,64,256]
    float* o_cls    = out + 1024 + BB * KK * HD; // [8]
    float* o_tidx   = o_cls + BB;                // [8,12]
    float* o_tsc    = o_tidx + BB * TOPKN;       // [8,12]

    // bump arena in d_ws
    char* base = (char*)d_ws;
    size_t off = 0;
    auto alloc = [&](size_t nfloats) -> float* {
        float* p = (float*)(base + off);
        off += nfloats * sizeof(float);
        off = (off + 255) & ~(size_t)255;
        return p;
    };

    float* regions = alloc((size_t)BB * KK * 1024); // concat [512,1024]
    float* gw      = alloc((size_t)BB * HD);        // graph feat [8,256]
    const size_t mark = off;

    auto gemm = [&](const float* A, const float* W, const float* bias, float* C,
                    int M, int N, int Kd, int lda, int ldc, int act) {
        dim3 g((N + BN - 1) / BN, (M + BM - 1) / BM);
        gemm_bf16_wmma<<<g, 256, 0, stream>>>(A, W, bias, C, M, N, Kd, lda, ldc, act);
    };
    auto fillf = [&](float* p, float v, long n) {
        fill_f<<<(unsigned)((n + 255) / 256), 256, 0, stream>>>(p, v, n);
    };
    auto ln = [&](const float* x, const float* res, float* o, const float* g,
                  const float* b, int rows, int D, int pre, int post) {
        row_ln<<<rows, D, 0, stream>>>(x, res, o, g, b, D, pre, post);
    };

    // ======================= AST: 3x GCN + concat MLP =======================
    {
        float* deg  = alloc(NAST);
        float* xw   = alloc((size_t)NAST * HD);
        float* tA   = alloc((size_t)NAST * HD);
        float* tB   = alloc((size_t)NAST * HD);
        float* hier = alloc((size_t)NAST * 768);

        auto gcn = [&](const float* xin, const int* ei, int E, const float* W,
                       const float* bias, float* ob) {
            fillf(deg, 1.f, NAST);
            deg_accum<<<(E + 255) / 256, 256, 0, stream>>>(ei + E, E, deg);
            dinv_finish<<<(NAST + 255) / 256, 256, 0, stream>>>(deg, NAST);
            gemm(xin, W, nullptr, xw, NAST, HD, HD, HD, HD, ACT_NONE);
            long tot = (long)NAST * HD;
            gcn_self<<<(unsigned)((tot + 255) / 256), 256, 0, stream>>>(xw, deg, bias, ob, tot);
            gcn_scatter<<<E, HD, 0, stream>>>(xw, ei, ei + E, deg, ob);
        };

        gcn(PF(IN_AST_X), PI(IN_TOKEN_EI), E_TOK, PF(P_AST_TW), PF(P_AST_TB), tA);
        ln(tA, nullptr, tA, PF(P_AST_TG), PF(P_AST_TBT), NAST, HD, ACT_RELU, ACT_NONE);
        copy_block<<<(unsigned)(((long)NAST * HD + 255) / 256), 256, 0, stream>>>(hier, 768, 0, tA, HD, (long)NAST * HD);

        gcn(tA, PI(IN_STMT_EI), E_STM, PF(P_AST_SW), PF(P_AST_SB), tB);
        ln(tB, nullptr, tB, PF(P_AST_SG), PF(P_AST_SBT), NAST, HD, ACT_RELU, ACT_NONE);
        copy_block<<<(unsigned)(((long)NAST * HD + 255) / 256), 256, 0, stream>>>(hier, 768, 256, tB, HD, (long)NAST * HD);

        gcn(tB, PI(IN_BLOCK_EI), E_BLK, PF(P_AST_BW), PF(P_AST_BB), tA);
        ln(tA, nullptr, tA, PF(P_AST_BG), PF(P_AST_BBT), NAST, HD, ACT_RELU, ACT_NONE);
        copy_block<<<(unsigned)(((long)NAST * HD + 255) / 256), 256, 0, stream>>>(hier, 768, 512, tA, HD, (long)NAST * HD);

        gemm(hier, PF(P_AST_O1W), PF(P_AST_O1B), tB, NAST, HD, 768, 768, HD, ACT_GELU);
        gemm(tB, PF(P_AST_O2W), PF(P_AST_O2B), tA, NAST, HD, HD, HD, HD, ACT_NONE);
        region_pool<<<BB * KK, HD, 0, stream>>>(tA, PI(IN_R_AST), PB(IN_R_AST_M), regions, MM, 1024, 0);
    }
    off = mark;

    // ======================= CFG: GGNN ======================================
    {
        float* h1 = alloc((size_t)NCFG * HD);
        float* h2 = alloc((size_t)NCFG * HD);
        float* hw = alloc((size_t)NCFG * HD);
        float* m  = alloc((size_t)NCFG * HD);
        float* gi = alloc((size_t)NCFG * 768);
        float* gh = alloc((size_t)NCFG * 768);
        long tot = (long)NCFG * HD;
        copy_f<<<(unsigned)((tot + 255) / 256), 256, 0, stream>>>(h1, PF(IN_CFG_X), tot);
        float* hc = h1; float* hn = h2;
        for (int t = 0; t < GSTEPS; ++t) {
            gemm(hc, PF(P_CFG_W) + (size_t)t * HD * HD, nullptr, hw, NCFG, HD, HD, HD, HD, ACT_NONE);
            fillf(m, 0.f, tot);
            seg_sum_scatter<<<E_CFG, HD, 0, stream>>>(hw, PI(IN_CFG_EI), PI(IN_CFG_EI) + E_CFG, m);
            gemm(m, PF(P_CFG_WIH), PF(P_CFG_BIH), gi, NCFG, 768, HD, HD, 768, ACT_NONE);
            gemm(hc, PF(P_CFG_WHH), PF(P_CFG_BHH), gh, NCFG, 768, HD, HD, 768, ACT_NONE);
            gru_gate<<<(unsigned)((tot + 255) / 256), 256, 0, stream>>>(gi, gh, hc, hn, tot);
            float* tmp = hc; hc = hn; hn = tmp;
        }
        ln(hc, nullptr, hc, PF(P_CFG_G), PF(P_CFG_B), NCFG, HD, ACT_RELU, ACT_NONE);
        region_pool<<<BB * KK, HD, 0, stream>>>(hc, PI(IN_R_CFG), PB(IN_R_CFG_M), regions, MM, 1024, 256);
    }
    off = mark;

    // ======================= PDG: 2x GAT ====================================
    {
        float* p1 = alloc((size_t)NPDG * HD);
        float* p2 = alloc((size_t)NPDG * HD);
        float* hh = alloc((size_t)NPDG * HD);
        float* a_s = alloc((size_t)NPDG * NHEADS);
        float* a_d = alloc((size_t)NPDG * NHEADS);
        unsigned* mx = (unsigned*)alloc((size_t)NPDG * NHEADS);
        float* den = alloc((size_t)NPDG * NHEADS);
        float* ebuf = alloc((size_t)(E_PDG + NPDG) * NHEADS);
        const int* src = PI(IN_PDG_EI);
        const int* dst = PI(IN_PDG_EI) + E_PDG;
        long ne = (long)(E_PDG + NPDG) * NHEADS;
        long nt = (long)NPDG * HD;

        auto gat = [&](const float* xin, const float* W, const float* aS, const float* aD,
                       const float* bias, const float* lg, const float* lb, float* ob) {
            gemm(xin, W, nullptr, hh, NPDG, HD, HD, HD, HD, ACT_NONE);
            gat_coef<<<(NPDG * NHEADS + 255) / 256, 256, 0, stream>>>(hh, aS, aD, a_s, a_d, NPDG);
            fill_u<<<(unsigned)((NPDG * NHEADS + 255) / 256), 256, 0, stream>>>(mx, 0u, NPDG * NHEADS);
            gat_edge_max<<<(unsigned)((ne + 255) / 256), 256, 0, stream>>>(src, dst, a_s, a_d, mx, E_PDG, NPDG);
            fillf(den, 0.f, NPDG * NHEADS);
            gat_edge_exp<<<(unsigned)((ne + 255) / 256), 256, 0, stream>>>(src, dst, a_s, a_d, mx, ebuf, den, E_PDG, NPDG);
            init_rows_bias<<<(unsigned)((nt + 255) / 256), 256, 0, stream>>>(ob, bias, nt, HD);
            gat_scatter<<<(unsigned)((ne * HC + 255) / 256), 256, 0, stream>>>(hh, src, dst, ebuf, den, ob, E_PDG, NPDG);
            ln(ob, nullptr, ob, lg, lb, NPDG, HD, ACT_ELU, ACT_NONE);
        };
        gat(PF(IN_PDG_X), PF(P_PDG_G1_W), PF(P_PDG_G1_AS), PF(P_PDG_G1_AD),
            PF(P_PDG_G1_BIAS), PF(P_PDG_LN1_G), PF(P_PDG_LN1_B), p1);
        gat(p1, PF(P_PDG_G2_W), PF(P_PDG_G2_AS), PF(P_PDG_G2_AD),
            PF(P_PDG_G2_BIAS), PF(P_PDG_LN2_G), PF(P_PDG_LN2_B), p2);
        region_pool<<<BB * KK, HD, 0, stream>>>(p2, PI(IN_R_PDG), PB(IN_R_PDG_M), regions, MM, 1024, 512);
    }
    off = mark;

    // ======================= SEM: 3x transformer layer ======================
    {
        const int R = BB * LL; // 4096 rows
        float* x    = alloc((size_t)R * HD);
        float* qkv  = alloc((size_t)R * 768);
        float* att  = alloc((size_t)R * HD);
        float* proj = alloc((size_t)R * HD);
        float* ffn  = alloc((size_t)R * 1024);
        float* x1   = alloc((size_t)R * HD);
        int*   pad  = (int*)alloc(R);
        long tot = (long)R * HD;
        copy_f<<<(unsigned)((tot + 255) / 256), 256, 0, stream>>>(x, PF(IN_LINES), tot);
        pad_mask_kernel<<<(R + 255) / 256, 256, 0, stream>>>(PF(IN_LINES), pad, R);
        pad_fix_kernel<<<BB, 32, 0, stream>>>(pad, LL);
        for (int l = 0; l < 3; ++l) {
            const int pb = P_SEM0 + l * 12;
            gemm(x, PF(pb + SWQKV), PF(pb + SBQKV), qkv, R, 768, HD, HD, 768, ACT_NONE);
            attn_kernel<<<dim3(LL, NHEADS, BB), 128, 0, stream>>>(qkv, pad, att, LL);
            gemm(att, PF(pb + SWO), PF(pb + SBO), proj, R, HD, HD, HD, HD, ACT_NONE);
            ln(proj, x, x1, PF(pb + SLN1G), PF(pb + SLN1B), R, HD, ACT_NONE, ACT_NONE);
            gemm(x1, PF(pb + SW1), PF(pb + SB1), ffn, R, 1024, HD, HD, 1024, ACT_GELU);
            gemm(ffn, PF(pb + SW2), PF(pb + SB2), proj, R, HD, 1024, 1024, HD, ACT_NONE);
            ln(proj, x1, x, PF(pb + SLN2G), PF(pb + SLN2B), R, HD, ACT_NONE, ACT_NONE);
        }
        line_pool<<<BB * KK, HD, 0, stream>>>(x, PI(IN_R_LINE), PB(IN_R_LINE_M), regions, KK, LNL, LL, 1024, 768);
    }
    off = mark;

    // ======================= Fuse + heads + topk ============================
    {
        const int R = BB * KK; // 512
        float* ftmp = alloc((size_t)R * HD);
        float* s1   = alloc((size_t)R * HD);
        float* s2   = alloc((size_t)R * 64);
        float* c1   = alloc((size_t)BB * HD);
        float* c2   = alloc((size_t)BB * 64);

        gemm(regions, PF(P_FUSE_W), PF(P_FUSE_B), ftmp, R, HD, 1024, 1024, HD, ACT_NONE);
        ln(ftmp, nullptr, o_feat, PF(P_FUSE_G), PF(P_FUSE_BT), R, HD, ACT_GELU, ACT_NONE);

        // scorer head
        gemm(o_feat, PF(P_SC_W1), PF(P_SC_B1), s1, R, HD, HD, HD, HD, ACT_NONE);
        ln(s1, nullptr, s1, PF(P_SC_G1), PF(P_SC_BT1), R, HD, ACT_NONE, ACT_GELU);
        gemm(s1, PF(P_SC_W2), PF(P_SC_B2), s2, R, 64, HD, HD, 64, ACT_NONE);
        ln(s2, nullptr, s2, PF(P_SC_G2), PF(P_SC_BT2), R, 64, ACT_NONE, ACT_GELU);
        dot64<<<R, 64, 0, stream>>>(s2, PF(P_SC_W3), PF(P_SC_B3), o_logits, R);

        mask_to_float<<<(R + 255) / 256, 256, 0, stream>>>(PB(IN_R_MASK), o_mask, R);
        topk_kernel<<<BB, HD, 0, stream>>>(o_feat, o_logits, PB(IN_R_MASK), gw, o_tidx, o_tsc);

        // classifier head
        gemm(gw, PF(P_CLS_W1), PF(P_CLS_B1), c1, BB, HD, HD, HD, HD, ACT_NONE);
        ln(c1, nullptr, c1, PF(P_CLS_G1), PF(P_CLS_BT1), BB, HD, ACT_NONE, ACT_GELU);
        gemm(c1, PF(P_CLS_W2), PF(P_CLS_B2), c2, BB, 64, HD, HD, 64, ACT_NONE);
        ln(c2, nullptr, c2, PF(P_CLS_G2), PF(P_CLS_BT2), BB, 64, ACT_NONE, ACT_GELU);
        dot64<<<BB, 64, 0, stream>>>(c2, PF(P_CLS_W3), PF(P_CLS_B3), o_cls, BB);
    }
}
